// DeepWalk_49855980372427
// MI455X (gfx1250) — compile-verified, matrix operand order checked by reading, not run on hardware
//
#include <hip/hip_runtime.h>
#include <hip/hip_bf16.h>

typedef __attribute__((ext_vector_type(16))) _Float16 v16h;
typedef __attribute__((ext_vector_type(4)))  _Float16 v4h;
typedef __attribute__((ext_vector_type(8)))  float    v8f;
typedef __attribute__((ext_vector_type(4)))  unsigned int v4u;

#define WALK   40
#define PAIRS  370
#define NEGS   5
#define DIM    128
#define BATCH  512
#define NPOS   (BATCH * PAIRS)          // 189440
#define NNEG   (BATCH * PAIRS * NEGS)   // 947200
#define NEGQ   (PAIRS * NEGS)           // 1850 per batch element

// softplus(x) = log(1 + e^x) for x in [-6, 6] (post-clip). Inputs to the log
// are in [1.0025, 404]: no denorm/inf handling needed, so use the raw
// hardware transcendentals (v_exp_f32 / v_log_f32 are base-2).
__device__ __forceinline__ float softplus_clipped(float x) {
    float t = __builtin_amdgcn_exp2f(x * 1.44269504f);   // 2^(x*log2e) = e^x
    return 0.69314718f * __builtin_amdgcn_logf(1.0f + t);
}

__device__ __forceinline__ float clip6(float s) {
    return fminf(6.0f, fmaxf(-6.0f, s));
}

__global__ __launch_bounds__(256)
void dw_loss_kernel(const int* __restrict__ walk,      // [BATCH*WALK]
                    const int* __restrict__ negidx,    // [NNEG], values in [0, BATCH*WALK)
                    const float* __restrict__ node_emb,// [NUM_NODES*DIM]
                    const float* __restrict__ ctx_emb, // [NUM_NODES*DIM]
                    float* __restrict__ ws)            // [BATCH] partials
{
    // 48 padded rows (40 real + 8 zero) x 128 f16 per table
    __shared__ __align__(16) _Float16 sNode[48 * DIM];
    __shared__ __align__(16) _Float16 sCtx [48 * DIM];
    __shared__ int            sIds[48];
    __shared__ unsigned char  sDst[PAIRS + 2];
    __shared__ float          sRed[8];

    const int b    = blockIdx.x;
    const int tid  = threadIdx.x;
    const int lane = tid & 31;
    const int wave = tid >> 5;

    // ---- phase 0: walk ids + dst-of-pair table ----
    if (tid < 48) sIds[tid] = (tid < WALK) ? walk[b * WALK + tid] : 0;
    if (tid == 0) {
        int q = 0;
        for (int i = 0; i < WALK; ++i) {
            int lo = (i - 5 > 0) ? i - 5 : 0;
            for (int j = lo; j < i; ++j) sDst[q++] = (unsigned char)i;
            int hi = (i + 6 < WALK) ? i + 6 : WALK;
            for (int j = i + 1; j < hi; ++j) sDst[q++] = (unsigned char)i;
        }
    }
    __syncthreads();

    // ---- phase 1: gather 80 rows from HBM, convert f32->f16 into LDS ----
    for (int rr = wave; rr < 96; rr += 8) {
        const bool isCtx = rr >= 48;
        const int  rl    = isCtx ? rr - 48 : rr;
        _Float16* dp = (isCtx ? sCtx : sNode) + rl * DIM + lane * 4;
        if (rl < WALK) {
            const float* sp = (isCtx ? ctx_emb : node_emb)
                              + (long long)sIds[rl] * DIM + lane * 4;
            float4 v = *(const float4*)sp;
            v4h hv;
            hv[0] = (_Float16)v.x; hv[1] = (_Float16)v.y;
            hv[2] = (_Float16)v.z; hv[3] = (_Float16)v.w;
            *(v4h*)dp = hv;
        } else {
            v4h z = {};
            *(v4h*)dp = z;
        }
    }
    __syncthreads();

    float pos_acc = 0.0f;
    float neg_acc = 0.0f;

    // ---- phase 2: positive pairs via WMMA Gram tiles ----
    // 9 tiles of the 48x48 (padded) Gram matrix; K = 128 = 4 x 32
    const int h    = lane >> 4;   // K-half selector per ISA A/B layout
    const int rsel = lane & 15;   // row (A) / column (B) within tile
    for (int tile = wave; tile < 9; tile += 8) {
        const int mt = tile / 3, nt = tile % 3;
        v8f acc = {};
#pragma unroll
        for (int kb = 0; kb < 4; ++kb) {
            const v4u* pa = (const v4u*)(sNode + (mt * 16 + rsel) * DIM + kb * 32 + h * 8);
            const v4u* pb = (const v4u*)(sCtx  + (nt * 16 + rsel) * DIM + kb * 32 + h * 8);
            v16h a, bm;
            ((v4u*)&a)[0]  = pa[0]; ((v4u*)&a)[1]  = pa[2];  // +0 and +32 bytes
            ((v4u*)&bm)[0] = pb[0]; ((v4u*)&bm)[1] = pb[2];
            acc = __builtin_amdgcn_wmma_f32_16x16x32_f16(
                false, a, false, bm, (short)0, acc, false, false);
        }
#pragma unroll
        for (int r = 0; r < 8; ++r) {
            const int m = mt * 16 + r + 8 * h;   // C/D layout: lanes16-31 hold M=r+8
            const int n = nt * 16 + rsel;
            int d = m - n; if (d < 0) d = -d;
            if (m < WALK && n < WALK && d >= 1 && d <= 5) {
                pos_acc += softplus_clipped(-clip6(acc[r]));  // -log_sigmoid(s)
            }
        }
    }

    // ---- phase 3: negatives, 2 pairs per wave (16 lanes / 8 floats each) ----
    const int g   = lane >> 4;    // pair-group within wave (0/1)
    const int sub = lane & 15;    // lane within group: covers k = sub*8 .. +7
    for (int q0 = wave * 2; q0 < NEGQ; q0 += 16) {
        const int  q     = q0 + g;
        const bool valid = q < NEGQ;
        const int  qs    = valid ? q : (NEGQ - 1);
        const int  p     = qs / NEGS;
        const int  srow  = sDst[p];                    // local node row (dst of pair)
        const int  fl    = negidx[b * NEGQ + qs];      // flat index into batch walk
        const long long nid = (long long)walk[fl];
        const float* crow = ctx_emb + nid * DIM;

        float s = 0.0f;
#pragma unroll
        for (int j = 0; j < 2; ++j) {
            const int c = j ^ g;                       // stagger chunks: group 0/1 hit
            const int k = sub * 8 + c * 4;             // disjoint LDS banks per ds op
            const float4 cv = *(const float4*)(crow + k);
            const v4h    nh = *(const v4h*)(sNode + srow * DIM + k);
            s += cv.x * (float)nh[0] + cv.y * (float)nh[1]
               + cv.z * (float)nh[2] + cv.w * (float)nh[3];
        }
#pragma unroll
        for (int o = 8; o > 0; o >>= 1) s += __shfl_xor(s, o, 32);  // within 16-lane group
        if (sub == 0 && valid) {
            neg_acc += softplus_clipped(clip6(s));     // -log_sigmoid(-s)
        }
    }

    // ---- block reduction of scaled partial ----
    float part = pos_acc * (1.0f / (float)NPOS)
               + neg_acc * ((float)NEGS / (float)NNEG);
#pragma unroll
    for (int o = 16; o > 0; o >>= 1) part += __shfl_xor(part, o, 32);
    if (lane == 0) sRed[wave] = part;
    __syncthreads();
    if (tid == 0) {
        float t = 0.0f;
        for (int w = 0; w < 8; ++w) t += sRed[w];
        ws[b] = t;
    }
}

__global__ __launch_bounds__(512)
void dw_reduce_kernel(const float* __restrict__ ws, float* __restrict__ out)
{
    __shared__ float sm[16];
    const int tid  = threadIdx.x;
    const int lane = tid & 31;
    const int wave = tid >> 5;
    float v = ws[tid];
#pragma unroll
    for (int o = 16; o > 0; o >>= 1) v += __shfl_xor(v, o, 32);
    if (lane == 0) sm[wave] = v;
    __syncthreads();
    if (tid == 0) {
        float t = 0.0f;
        for (int w = 0; w < 16; ++w) t += sm[w];
        out[0] = t;
    }
}

extern "C" void kernel_launch(void* const* d_in, const int* in_sizes, int n_in,
                              void* d_out, int out_size, void* d_ws, size_t ws_size,
                              hipStream_t stream) {
    const int*   walk     = (const int*)d_in[0];   // batch_walk [512*40]
    const int*   negidx   = (const int*)d_in[1];   // neg_idx_dst [947200]
    const float* node_emb = (const float*)d_in[2]; // node_embed [1e6*128]
    const float* ctx_emb  = (const float*)d_in[3]; // context_embed [1e6*128]
    float*       partials = (float*)d_ws;          // 512 floats scratch
    float*       out      = (float*)d_out;

    dw_loss_kernel<<<BATCH, 256, 0, stream>>>(walk, negidx, node_emb, ctx_emb, partials);
    dw_reduce_kernel<<<1, 512, 0, stream>>>(partials, out);
}
